// SDE_75299366633585
// MI455X (gfx1250) — matrix operand order repaired and checked by reading, not compiled, not run on hardware
//
#include <hip/hip_runtime.h>

typedef __attribute__((ext_vector_type(16))) _Float16 v16h;
typedef __attribute__((ext_vector_type(8)))  _Float16 v8h;
typedef __attribute__((ext_vector_type(8)))  float    v8f;
typedef __attribute__((ext_vector_type(4)))  int      int4v;

#define C_      256
#define T_      64
#define V_      25
#define TV_     1600          // T*V (stride of c within state layout)
#define M_TOT   25600         // B*T*V rows
#define S_ELEMS 6553600       // B*C*T*V = 16*256*64*25
#define NSTEP   8

#if defined(__AMDGCN__) && __has_builtin(__builtin_amdgcn_global_load_async_to_lds_b32)
#define HAVE_ASYNC_LDS 1
#else
#define HAVE_ASYNC_LDS 0
#endif

#if defined(__AMDGCN__) && __has_builtin(__builtin_amdgcn_global_load_async_to_lds_b128)
#define HAVE_ASYNC_LDS_B128 1
#else
#define HAVE_ASYNC_LDS_B128 0
#endif

typedef __attribute__((address_space(1))) int   g_int;
typedef __attribute__((address_space(3))) int   l_int;
typedef __attribute__((address_space(1))) int4v g_int4;
typedef __attribute__((address_space(3))) int4v l_int4;

__device__ __forceinline__ void wait_async_then_barrier() {
#if HAVE_ASYNC_LDS
#if __has_builtin(__builtin_amdgcn_s_wait_asynccnt)
    __builtin_amdgcn_s_wait_asynccnt(0);
#else
    asm volatile("s_wait_asynccnt 0" ::: "memory");
#endif
#endif
    __syncthreads();
}

// ---------------- deterministic counter-based RNG ----------------
__device__ __forceinline__ unsigned long long smix64(unsigned long long x) {
    x += 0x9E3779B97F4A7C15ull;
    x = (x ^ (x >> 30)) * 0xBF58476D1CE4E5B9ull;
    x = (x ^ (x >> 27)) * 0x94D049BB133111EBull;
    return x ^ (x >> 31);
}

__device__ __forceinline__ float gauss_from_u64(unsigned long long h) {
    unsigned int u1 = (unsigned int)h;
    unsigned int u2 = (unsigned int)(h >> 32);
    float f1 = ((float)u1 + 1.0f) * 2.3283064365386963e-10f;   // (0,1]
    float f2 = (float)u2 * 2.3283064365386963e-10f;            // [0,1)
    float rad = __fsqrt_rn(-2.0f * __logf(f1));
    return rad * __cosf(6.283185307179586f * f2);
}

// ---------------- W (f32, [c][d]) -> WhT (f16, [d][c]) ----------------
__global__ void wconv_kernel(const float* __restrict__ Wmu,
                             const float* __restrict__ Wsg,
                             _Float16* __restrict__ WhTmu,
                             _Float16* __restrict__ WhTsg) {
    int i = blockIdx.x * blockDim.x + threadIdx.x;   // 0..65535
    int d = i >> 8;
    int c = i & 255;
    WhTmu[i] = (_Float16)Wmu[c * C_ + d];
    WhTsg[i] = (_Float16)Wsg[c * C_ + d];
}

// ---------------- z (f32, bctv) -> stateT0 (f16, [m][c]) ----------------
__global__ void zt_init_kernel(const float* __restrict__ z,
                               _Float16* __restrict__ stateT0) {
    int j = blockIdx.x * blockDim.x + threadIdx.x;   // < M_TOT * C_
    int m = j >> 8;
    int c = j & 255;
    int b = m / TV_;
    int q = m / V_;
    int t = q & (T_ - 1);
    int v = m - q * V_;
    long idx = (((long)b * C_ + c) * T_ + t) * V_ + v;
    stateT0[j] = (_Float16)z[idx];
}

// ---------------- slot-0 copy: out_shift[0]=z, out_cls[0]=z ----------------
__global__ void copy0_kernel(const float4* __restrict__ z,
                             float4* __restrict__ cls0,
                             float4* __restrict__ shf0) {
    int i = blockIdx.x * blockDim.x + threadIdx.x;   // S_ELEMS/4 threads
    float4 x = z[i];
    cls0[i] = x;
    shf0[i] = x;
}

// ---------------- one Milstein step: GEMM(WMMA) + pointwise ----------------
// grid: (1600, 2) blocks of 256 threads (8 waves).
// block covers 16 M-rows x 128 D-cols; wave w owns a 16x16 D-tile.
// FAST: A-tile comes from the f16 transposed state copy (contiguous async
//       b128 DMA into LDS); epilogue also writes the next f16 state copy.
// !FAST: A-tile gathered f32 (async b32) + one-shot LDS f32->f16 pass.
template <bool FAST>
__global__ void __launch_bounds__(256)
sde_step_kernel(const float* __restrict__ state_in,    // cls slot (step-1), f32
                float* __restrict__ cls_out,           // cls slot step
                float* __restrict__ shift_out,         // shift slot step
                const _Float16* __restrict__ stateT_in,// f16 [m][c] (FAST)
                _Float16* __restrict__ stateT_out,     // f16 [m][c] (FAST)
                const _Float16* __restrict__ WhTmu,    // [d][c] f16
                const _Float16* __restrict__ WhTsg,    // [d][c] f16
                const float* __restrict__ bmu,
                const float* __restrict__ bsg,
                int step) {
    __shared__ __align__(16) _Float16 Ash[16 * C_];   // 8 KB f16 A tile

    const int tid  = threadIdx.x;
    const int lane = tid & 31;
    const int wave = tid >> 5;

    if constexpr (FAST) {
        // contiguous 8 KB block copy: stateT rows [tile*16, tile*16+16)
        const _Float16* src = stateT_in + (long)blockIdx.x * (16 * C_);
#pragma unroll
        for (int it = 0; it < 2; ++it) {
            int off = (tid + it * 256) * 8;            // in halfs, 16B chunks
#if HAVE_ASYNC_LDS_B128
            __builtin_amdgcn_global_load_async_to_lds_b128(
                (g_int4*)(src + off), (l_int4*)&Ash[off], 0, 0);
#elif HAVE_ASYNC_LDS
#pragma unroll
            for (int w2 = 0; w2 < 4; ++w2)
                __builtin_amdgcn_global_load_async_to_lds_b32(
                    (g_int*)(src + off + w2 * 2), (l_int*)&Ash[off + w2 * 2], 0, 0);
#else
            *reinterpret_cast<v8h*>(&Ash[off]) =
                *reinterpret_cast<const v8h*>(src + off);
#endif
        }
        wait_async_then_barrier();
    } else {
        __shared__ float Asf[16 * C_];                 // 16 KB f32 gather tile
        {
            int r  = tid >> 4;
            int c0 = tid & 15;
            int m  = blockIdx.x * 16 + r;
            int b  = m / TV_;
            int t  = (m / V_) & (T_ - 1);
            int v  = m - (m / V_) * V_;
            long rowbase = (((long)b * C_) * T_ + t) * V_ + v;
#pragma unroll
            for (int it = 0; it < 16; ++it) {
                int c = c0 + it * 16;
#if HAVE_ASYNC_LDS
                __builtin_amdgcn_global_load_async_to_lds_b32(
                    (g_int*)(state_in + rowbase + (long)c * TV_),
                    (l_int*)&Asf[r * C_ + c], 0, 0);
#else
                Asf[r * C_ + c] = state_in[rowbase + (long)c * TV_];
#endif
            }
        }
        wait_async_then_barrier();
        // one-shot f32 -> f16 conversion pass (off the K-loop critical path)
        {
            int r  = tid >> 4;
            int cb = (tid & 15) * 16;
            const float* s = &Asf[r * C_ + cb];
            float4 q0 = *reinterpret_cast<const float4*>(s);
            float4 q1 = *reinterpret_cast<const float4*>(s + 4);
            float4 q2 = *reinterpret_cast<const float4*>(s + 8);
            float4 q3 = *reinterpret_cast<const float4*>(s + 12);
            v8h h0 = {(_Float16)q0.x, (_Float16)q0.y, (_Float16)q0.z, (_Float16)q0.w,
                      (_Float16)q1.x, (_Float16)q1.y, (_Float16)q1.z, (_Float16)q1.w};
            v8h h1 = {(_Float16)q2.x, (_Float16)q2.y, (_Float16)q2.z, (_Float16)q2.w,
                      (_Float16)q3.x, (_Float16)q3.y, (_Float16)q3.z, (_Float16)q3.w};
            *reinterpret_cast<v8h*>(&Ash[r * C_ + cb])     = h0;
            *reinterpret_cast<v8h*>(&Ash[r * C_ + cb + 8]) = h1;
        }
        __syncthreads();
    }

    // ---- WMMA K-loop (pure f16, no conversions) ----
    const int N   = lane & 15;       // D column within tile (C/D layout)
    const int hi  = lane >> 4;       // half-wave selector
    const int dtile = blockIdx.y * 128 + wave * 16;
    const int d   = dtile + N;

    const _Float16* __restrict__ Bmu  = WhTmu + (long)d * C_;
    const _Float16* __restrict__ Bsg  = WhTsg + (long)d * C_;
    const _Float16* __restrict__ Arow = &Ash[(lane & 15) * C_];

    v8f accMu = {};
    v8f accSg = {};

    union Frag { v16h v; v8h h[2]; };

#pragma unroll
    for (int kc = 0; kc < C_; kc += 32) {
        __builtin_prefetch((const void*)(Bmu + kc + 64), 0, 3);
        __builtin_prefetch((const void*)(Bsg + kc + 64), 0, 3);

        // A fragment: 16 contiguous halves (K = kc + 16*hi + e), M = lane&15
        Frag a;
        a.h[0] = *reinterpret_cast<const v8h*>(Arow + kc + 16 * hi);
        a.h[1] = *reinterpret_cast<const v8h*>(Arow + kc + 16 * hi + 8);

        // B fragments: 16 contiguous halves (K = kc + 16*hi + e), N = lane&15
        Frag bm, bs;
        bm.h[0] = *reinterpret_cast<const v8h*>(Bmu + kc + 16 * hi);
        bm.h[1] = *reinterpret_cast<const v8h*>(Bmu + kc + 16 * hi + 8);
        bs.h[0] = *reinterpret_cast<const v8h*>(Bsg + kc + 16 * hi);
        bs.h[1] = *reinterpret_cast<const v8h*>(Bsg + kc + 16 * hi + 8);

        accMu = __builtin_amdgcn_wmma_f32_16x16x32_f16(
                    false, a.v, false, bm.v, (short)0, accMu, false, false);
        accSg = __builtin_amdgcn_wmma_f32_16x16x32_f16(
                    false, a.v, false, bs.v, (short)0, accSg, false, false);
    }

    // ---- pointwise Milstein update + stores ----
    const float biasMu = bmu[d];
    const float biasSg = bsg[d];
    const unsigned long long keyA = smix64(0xA5A5ull * (unsigned)(2 * step + 0) + 0x1234567ull);
    const unsigned long long keyB = smix64(0xA5A5ull * (unsigned)(2 * step + 1) + 0x89ABCDFull);

#pragma unroll
    for (int rr = 0; rr < 8; ++rr) {
        int Mrow = rr + hi * 8;                 // C/D 16x16 f32 layout
        int m = blockIdx.x * 16 + Mrow;
        int b = m / TV_;
        int q = m / V_;
        int t = q & (T_ - 1);
        int v = m - q * V_;
        long idx = (((long)b * C_ + d) * T_ + t) * V_ + v;

        float zold = state_in[idx];
        float mu = accMu[rr] + biasMu;
        float sr = accSg[rr] + biasSg;
        float s  = (sr > 20.0f) ? sr : __logf(1.0f + __expf(sr));  // softplus

        unsigned long long e = ((unsigned long long)m << 32) | (unsigned)d;
        float dw1 = gauss_from_u64(smix64(e ^ keyA));
        float dw2 = gauss_from_u64(smix64(e ^ keyB));

        // DT = 1.0
        float zn = zold + mu + s * dw1 + 0.5f * s * s * zold * (dw2 * dw2 - 1.0f);

        cls_out[idx] = zn;                       // state snapshot (z4)
        int ts = t + step;                       // shift = step (DILATION=1)
        if (ts < T_) shift_out[idx + (long)step * V_] = zn;
        if (t < step) shift_out[idx] = 0.0f;     // zero-fill rolled-in region

        if constexpr (FAST) {                    // next step's A-layout state
            stateT_out[(long)m * C_ + d] = (_Float16)zn;
        }
    }
}

// ---------------- launch ----------------
extern "C" void kernel_launch(void* const* d_in, const int* in_sizes, int n_in,
                              void* d_out, int out_size, void* d_ws, size_t ws_size,
                              hipStream_t stream) {
    (void)in_sizes; (void)n_in; (void)out_size;

    const float* z    = (const float*)d_in[0];
    const float* Wmu  = (const float*)d_in[1];
    const float* bmu  = (const float*)d_in[2];
    const float* Wsg  = (const float*)d_in[3];
    const float* bsg  = (const float*)d_in[4];

    float* out        = (float*)d_out;
    float* shift_base = out;                                  // 9*S floats
    float* cls_base   = out + (size_t)(NSTEP + 1) * S_ELEMS;  // 9*S floats

    _Float16* WhTmu = (_Float16*)d_ws;
    _Float16* WhTsg = WhTmu + C_ * C_;

    const size_t w_bytes  = (size_t)2 * C_ * C_ * sizeof(_Float16);   // 256 KB
    const size_t st_halfs = (size_t)M_TOT * C_;                       // 6.55M
    const size_t need     = w_bytes + 2 * st_halfs * sizeof(_Float16);
    const bool   fast     = (ws_size >= need);

    _Float16* stateT0 = (_Float16*)((char*)d_ws + w_bytes);
    _Float16* stateT1 = stateT0 + st_halfs;

    wconv_kernel<<<256, 256, 0, stream>>>(Wmu, Wsg, WhTmu, WhTsg);
    copy0_kernel<<<S_ELEMS / 4 / 256, 256, 0, stream>>>(
        (const float4*)z, (float4*)cls_base, (float4*)shift_base);
    if (fast) {
        zt_init_kernel<<<M_TOT, 256, 0, stream>>>(z, stateT0);
    }

    for (int step = 1; step <= NSTEP; ++step) {
        const float* st_in  = cls_base + (size_t)(step - 1) * S_ELEMS;
        float* cls_out      = cls_base + (size_t)step * S_ELEMS;
        float* shf_out      = shift_base + (size_t)step * S_ELEMS;
        if (fast) {
            _Float16* tin  = (step & 1) ? stateT0 : stateT1;
            _Float16* tout = (step & 1) ? stateT1 : stateT0;
            sde_step_kernel<true><<<dim3(1600, 2), 256, 0, stream>>>(
                st_in, cls_out, shf_out, tin, tout,
                WhTmu, WhTsg, bmu, bsg, step);
        } else {
            sde_step_kernel<false><<<dim3(1600, 2), 256, 0, stream>>>(
                st_in, cls_out, shf_out, nullptr, nullptr,
                WhTmu, WhTsg, bmu, bsg, step);
        }
    }
}
